// MambaLayer_6794638262863
// MI455X (gfx1250) — matrix operand order repaired and checked by reading, not compiled, not run on hardware
//
#include <hip/hip_runtime.h>
#include <hip/hip_bf16.h>
#include <cstddef>
#include <cstdint>

// ---------------------------------------------------------------------------
// Types for CDNA5 WMMA (wave32, 16x16x32 bf16 -> f32)
// ---------------------------------------------------------------------------
typedef __bf16 bf16_t;
typedef __attribute__((ext_vector_type(8)))  __bf16 v8bf;
typedef __attribute__((ext_vector_type(16))) __bf16 v16bf;
typedef __attribute__((ext_vector_type(8)))  float  v8f;

// Problem constants
constexpr int DIM     = 128;
constexpr int L       = 4096;      // 64*64
constexpr int DS      = 16;        // D_STATE
constexpr int DR      = 8;         // DT_RANK
constexpr int NPROJ   = 40;        // DT_RANK + 2*D_STATE
constexpr int NPROJP  = 64;        // padded
constexpr int NC      = 64;        // scan chunks
constexpr int CL      = 64;        // chunk length (NC*CL == L)
constexpr float EPS   = 1e-5f;

// ---------------------------------------------------------------------------
// Workspace layout (byte offsets, all 256B-aligned by construction)
// ---------------------------------------------------------------------------
constexpr size_t OFF_W1   = 0;                                   // nin_w  bf16 [128][128]
constexpr size_t OFF_W2   = OFF_W1   + (size_t)128*128*2;        // nin2_w bf16 [128][128]
constexpr size_t OFF_WTI  = OFF_W2   + (size_t)128*128*2;        // in_proj^T  bf16 [128][512]
constexpr size_t OFF_WXPT = OFF_WTI  + (size_t)128*512*2;        // x_proj^T   bf16 [256][64] (pad)
constexpr size_t OFF_WOT  = OFF_WXPT + (size_t)256*64*2;         // out_proj^T bf16 [256][128]
constexpr size_t OFF_AEXP = OFF_WOT  + (size_t)256*128*2;        // -exp(A_log) f32 [256][16]
constexpr size_t OFF_XB   = OFF_AEXP + (size_t)256*16*4;         // x bf16 [4][128][4096]
constexpr size_t OFF_H1   = OFF_XB   + (size_t)4*128*4096*2;     // h1 f32 [4][4096][128]
constexpr size_t OFF_XS   = OFF_H1   + (size_t)4*4096*128*4;     // xs bf16 [16][4096][128]
constexpr size_t OFF_XZ   = OFF_XS   + (size_t)16*4096*128*2;    // xz f32 [16][4096][512]
constexpr size_t OFF_UC   = OFF_XZ   + (size_t)16*4096*512*4;    // uc f32 [16][4096][256]
constexpr size_t OFF_UCB  = OFF_UC   + (size_t)16*4096*256*4;    // uc bf16
constexpr size_t OFF_DBL  = OFF_UCB  + (size_t)16*4096*256*2;    // dbl f32 [16][4096][40]
constexpr size_t OFF_YBF  = OFF_DBL  + (size_t)16*4096*40*4;     // gated y bf16 [16][4096][256]
constexpr size_t OFF_YM   = OFF_YBF  + (size_t)16*4096*256*2;    // ym f32 [16][4096][128]
constexpr size_t OFF_RT   = OFF_YM   + (size_t)16*4096*128*4;    // res bf16 [4][128][4096]
constexpr size_t OFF_P    = OFF_RT   + (size_t)4*128*4096*2;     // chunk decay   [16][64][256][16] f32
constexpr size_t OFF_S    = OFF_P    + (size_t)16*64*256*16*4;   // chunk partial [16][64][256][16] f32
constexpr size_t OFF_HIN  = OFF_S    + (size_t)16*64*256*16*4;   // chunk carry-in state

// ---------------------------------------------------------------------------
// Register-blocked 16x(16*NB) tile:  C = A(MxK,bf16) * Bt(KxN,bf16)
// A fragment loaded once per k-step, reused across NB WMMAs.
// Layouts per CDNA5 ISA 7.12.2:
//  A: lane = (K-half, M-row); 16 bf16/lane = 2 contiguous 16B chunks.
//  B: lane = (K-half, N-col); 16 bf16/lane gathered down K in A's k order.
// ---------------------------------------------------------------------------
template <int NB>
__device__ __forceinline__ void gemm_tile_bf16(const bf16_t* __restrict__ A, int lda,
                                               const bf16_t* __restrict__ Bt, int ldb,
                                               int m0, int n0, int K, int lane,
                                               v8f acc[NB])
{
  const int row  = lane & 15;
  const int half = lane >> 4;
  const bf16_t* pa = A  + (size_t)(m0 + row) * lda + half * 8;
  const bf16_t* pb = Bt + (size_t)(half * 8) * ldb + n0 + row;
  for (int k0 = 0; k0 < K; k0 += 32) {
    if (k0 + 32 < K) __builtin_prefetch(pa + k0 + 32, 0, 1);
    v8bf alo = *(const v8bf*)(pa + k0);
    v8bf ahi = *(const v8bf*)(pa + k0 + 16);
    v16bf av;
#pragma unroll
    for (int i = 0; i < 8; ++i) { av[i] = alo[i]; av[i + 8] = ahi[i]; }
    const bf16_t* pbk = pb + (size_t)k0 * ldb;
#pragma unroll
    for (int j = 0; j < NB; ++j) {
      v16bf bv;
#pragma unroll
      for (int i = 0; i < 8; ++i) { bv[i] = pbk[(size_t)i * ldb + j * 16]; bv[i + 8] = pbk[(size_t)(16 + i) * ldb + j * 16]; }
      acc[j] = __builtin_amdgcn_wmma_f32_16x16x32_bf16(false, av, false, bv,
                                                       (short)0, acc[j], false, false);
    }
  }
}

template <int NB>
__device__ __forceinline__ void zero_acc(v8f acc[NB])
{
#pragma unroll
  for (int j = 0; j < NB; ++j) acc[j] = v8f{0.f,0.f,0.f,0.f,0.f,0.f,0.f,0.f};
}

// ---------------------------------------------------------------------------
// K0: weight casts / transposes, x->bf16, A = -exp(A_log)
// ---------------------------------------------------------------------------
__global__ void prep_kernel(const float* __restrict__ x,
                            const float* __restrict__ nin_w,
                            const float* __restrict__ nin2_w,
                            const float* __restrict__ in_proj_w,
                            const float* __restrict__ x_proj_w,
                            const float* __restrict__ out_proj_w,
                            const float* __restrict__ A_log,
                            bf16_t* xb, bf16_t* w1, bf16_t* w2,
                            bf16_t* wti, bf16_t* wxpt, bf16_t* wot, float* Aexp)
{
  int i = blockIdx.x * 256 + threadIdx.x;
  constexpr int N0 = 4 * 128 * 4096;
  if (i < N0) { xb[i] = (bf16_t)x[i]; return; }
  i -= N0;
  if (i < 128 * 128) { w1[i] = (bf16_t)nin_w[i]; return; }
  i -= 128 * 128;
  if (i < 128 * 128) { w2[i] = (bf16_t)nin2_w[i]; return; }
  i -= 128 * 128;
  if (i < 128 * 512) { int c = i >> 9, o = i & 511; wti[i] = (bf16_t)in_proj_w[o * 128 + c]; return; }
  i -= 128 * 512;
  if (i < 256 * 64)  { int d = i >> 6, j = i & 63;
                       wxpt[i] = (bf16_t)((j < NPROJ) ? x_proj_w[j * 256 + d] : 0.f); return; }
  i -= 256 * 64;
  if (i < 256 * 128) { int d = i >> 7, o = i & 127; wot[i] = (bf16_t)out_proj_w[o * 256 + d]; return; }
  i -= 256 * 128;
  if (i < 256 * 16)  { Aexp[i] = -expf(A_log[i]); }
}

// ---------------------------------------------------------------------------
// K1: nin1 GEMM (per b: [128xL] = nin_w[128x128] * x[128xL]) + BN1 + ReLU
//     16x64 per wave, stores h1 as [b][l][c]
// ---------------------------------------------------------------------------
__global__ void nin1_gemm_kernel(const bf16_t* __restrict__ W, const bf16_t* __restrict__ Xb,
                                 const float* __restrict__ g,  const float* __restrict__ bt,
                                 const float* __restrict__ mn, const float* __restrict__ vr,
                                 float* __restrict__ h1)
{
  const int wave = (blockIdx.x * blockDim.x + threadIdx.x) >> 5;
  const int lane = threadIdx.x & 31;
  const int ng = wave & 63, mt = (wave >> 6) & 7, b = wave >> 9;   // 4*8*64 waves
  const int m0 = mt * 16, n0 = ng * 64;
  v8f acc[4]; zero_acc<4>(acc);
  gemm_tile_bf16<4>(W, 128, Xb + (size_t)b * 128 * L, L, m0, n0, 128, lane, acc);
  const int half = lane >> 4, col = lane & 15;
#pragma unroll
  for (int r = 0; r < 8; ++r) {
    const int o = m0 + r + 8 * half;
    const float sc = g[o] * rsqrtf(vr[o] + EPS);
    const float bi = bt[o] - mn[o] * sc;
#pragma unroll
    for (int j = 0; j < 4; ++j) {
      const int l = n0 + j * 16 + col;
      h1[((size_t)b * L + l) * 128 + o] = fmaxf(acc[j][r] * sc + bi, 0.f);
    }
  }
}

// ---------------------------------------------------------------------------
// K2: build 4 flipped direction copies, bf16:  xs[dir*4+b][l][c]
// ---------------------------------------------------------------------------
__global__ void build_xs_kernel(const float* __restrict__ h1, bf16_t* __restrict__ xs)
{
  const int idx = blockIdx.x * 256 + threadIdx.x;       // 16*4096*128
  const int c = idx & 127, l = (idx >> 7) & 4095, bb = idx >> 19;
  const int dir = bb >> 2, b = bb & 3;
  const int ls = (dir & 2) ? (L - 1 - l) : l;
  const int cs = (dir & 1) ? (127 - c)   : c;
  xs[idx] = (bf16_t)h1[((size_t)b * L + ls) * 128 + cs];
}

// ---------------------------------------------------------------------------
// K3: in_proj GEMM  xz[65536 x 512] = xs[65536 x 128] * wti[128 x 512]
// ---------------------------------------------------------------------------
__global__ void inproj_gemm_kernel(const bf16_t* __restrict__ xs, const bf16_t* __restrict__ wti,
                                   float* __restrict__ xz)
{
  const int wave = (blockIdx.x * blockDim.x + threadIdx.x) >> 5;
  const int lane = threadIdx.x & 31;
  const int ng = wave & 7, mt = wave >> 3;              // 4096 x 8 groups of 64
  const int m0 = mt * 16, n0 = ng * 64;
  v8f acc[4]; zero_acc<4>(acc);
  gemm_tile_bf16<4>(xs, 128, wti, 512, m0, n0, 128, lane, acc);
  const int half = lane >> 4, col = lane & 15;
#pragma unroll
  for (int r = 0; r < 8; ++r)
#pragma unroll
    for (int j = 0; j < 4; ++j)
      xz[(size_t)(m0 + r + 8 * half) * 512 + n0 + j * 16 + col] = acc[j][r];
}

// ---------------------------------------------------------------------------
// K4: causal depthwise conv (D_CONV=4) + SiLU on the u-half of xz
// ---------------------------------------------------------------------------
__global__ void conv_silu_kernel(const float* __restrict__ xz,
                                 const float* __restrict__ cw, const float* __restrict__ cb,
                                 float* __restrict__ uc, bf16_t* __restrict__ ucb)
{
  const int idx = blockIdx.x * 256 + threadIdx.x;       // 16*4096*256
  const int d = idx & 255, l = (idx >> 8) & 4095, bb = idx >> 20;
  float a = cb[d];
#pragma unroll
  for (int k = 0; k < 4; ++k) {
    const int ll = l + k - 3;
    if (ll >= 0) a += xz[((size_t)bb * L + ll) * 512 + d] * cw[d * 4 + k];
  }
  a = a / (1.f + __expf(-a));                           // SiLU
  uc[idx]  = a;
  ucb[idx] = (bf16_t)a;
}

// ---------------------------------------------------------------------------
// K5: x_proj GEMM  dbl[65536 x 40] = uc[65536 x 256] * wxpt[256 x 64(pad)]
// ---------------------------------------------------------------------------
__global__ void xproj_gemm_kernel(const bf16_t* __restrict__ ucb, const bf16_t* __restrict__ wxpt,
                                  float* __restrict__ dbl)
{
  const int wave = (blockIdx.x * blockDim.x + threadIdx.x) >> 5;
  const int lane = threadIdx.x & 31;
  const int mt = wave;                                  // 4096 M-tiles, one 64-wide group
  const int m0 = mt * 16;
  v8f acc[4]; zero_acc<4>(acc);
  gemm_tile_bf16<4>(ucb, 256, wxpt, NPROJP, m0, 0, 256, lane, acc);
  const int half = lane >> 4, col = lane & 15;
#pragma unroll
  for (int j = 0; j < 4; ++j) {
    const int n = j * 16 + col;
    if (n < NPROJ) {
#pragma unroll
      for (int r = 0; r < 8; ++r)
        dbl[(size_t)(m0 + r + 8 * half) * NPROJ + n] = acc[j][r];
    }
  }
}

// ---------------------------------------------------------------------------
// Selective scan, chunked (associative) formulation.
// Per (bb,d,n):  h_l = a_l*h_{l-1} + w_l,  a_l = exp(dt_l*A[d][n]),
//                w_l = dt_l*u_l*B_l[n].
// Pass 1: per chunk of CL=64 steps compute P = prod(a), S = local scan end.
// Pass 2: 64-step sequential scan over chunk summaries -> carry-in per chunk.
// Pass 3: re-run each chunk from its carry-in, emit gated y (bf16).
// ---------------------------------------------------------------------------
__global__ void scan_part1_kernel(const float* __restrict__ dbl, const float* __restrict__ uc,
                                  const float* __restrict__ Aexp,
                                  const float* __restrict__ wdt, const float* __restrict__ dtb,
                                  float* __restrict__ Pc, float* __restrict__ Sc)
{
  __shared__ float srow[DR + DS];                       // dtr(8) + B(16)
  const int bb = blockIdx.x >> 6;                       // 16
  const int ch = blockIdx.x & 63;                       // 64 chunks
  const int d  = threadIdx.x;
  float w[DR];
#pragma unroll
  for (int r = 0; r < DR; ++r) w[r] = wdt[d * DR + r];
  const float bias = dtb[d];
  float Ar[DS];
#pragma unroll
  for (int n = 0; n < DS; ++n) Ar[n] = Aexp[d * DS + n];
  float P[DS], S[DS];
#pragma unroll
  for (int n = 0; n < DS; ++n) { P[n] = 1.f; S[n] = 0.f; }

  const int l0 = ch * CL;
  for (int t = 0; t < CL; ++t) {
    const int l = l0 + t;
    __syncthreads();
    if (threadIdx.x < DR + DS)
      srow[threadIdx.x] = dbl[((size_t)bb * L + l) * NPROJ + threadIdx.x];
    __syncthreads();
    float dtv = bias;
#pragma unroll
    for (int r = 0; r < DR; ++r) dtv += srow[r] * w[r];
    dtv = (dtv > 20.f) ? dtv : log1pf(__expf(dtv));
    const float du = dtv * uc[((size_t)bb * L + l) * 256 + d];
#pragma unroll
    for (int n = 0; n < DS; ++n) {
      const float a = __expf(dtv * Ar[n]);
      S[n] = a * S[n] + du * srow[DR + n];
      P[n] *= a;
    }
  }
  const size_t base = (((size_t)bb * NC + ch) * 256 + d) * DS;
#pragma unroll
  for (int n = 0; n < DS; ++n) { Pc[base + n] = P[n]; Sc[base + n] = S[n]; }
}

__global__ void scan_part2_kernel(const float* __restrict__ Pc, const float* __restrict__ Sc,
                                  float* __restrict__ Hin)
{
  const int idx = blockIdx.x * 256 + threadIdx.x;       // 16*256*16
  const int n = idx & 15, d = (idx >> 4) & 255, bb = idx >> 12;
  float h = 0.f;
  for (int ch = 0; ch < NC; ++ch) {
    const size_t o = (((size_t)bb * NC + ch) * 256 + d) * DS + n;
    Hin[o] = h;
    h = Pc[o] * h + Sc[o];
  }
}

__global__ void scan_part3_kernel(const float* __restrict__ dbl, const float* __restrict__ uc,
                                  const float* __restrict__ xz,  const float* __restrict__ Aexp,
                                  const float* __restrict__ wdt, const float* __restrict__ dtb,
                                  const float* __restrict__ Dp,  const float* __restrict__ Hin,
                                  bf16_t* __restrict__ ybf)
{
  __shared__ float srow[NPROJ];                         // dtr(8) + B(16) + C(16)
  const int bb = blockIdx.x >> 6;
  const int ch = blockIdx.x & 63;
  const int d  = threadIdx.x;
  float w[DR];
#pragma unroll
  for (int r = 0; r < DR; ++r) w[r] = wdt[d * DR + r];
  const float bias = dtb[d];
  float Ar[DS];
#pragma unroll
  for (int n = 0; n < DS; ++n) Ar[n] = Aexp[d * DS + n];
  const float Dv = Dp[d];
  float h[DS];
  const size_t hbase = (((size_t)bb * NC + ch) * 256 + d) * DS;
#pragma unroll
  for (int n = 0; n < DS; ++n) h[n] = Hin[hbase + n];

  const int l0 = ch * CL;
  for (int t = 0; t < CL; ++t) {
    const int l = l0 + t;
    __syncthreads();
    if (threadIdx.x < NPROJ)
      srow[threadIdx.x] = dbl[((size_t)bb * L + l) * NPROJ + threadIdx.x];
    __syncthreads();
    float dtv = bias;
#pragma unroll
    for (int r = 0; r < DR; ++r) dtv += srow[r] * w[r];
    dtv = (dtv > 20.f) ? dtv : log1pf(__expf(dtv));
    const float u  = uc[((size_t)bb * L + l) * 256 + d];
    const float du = dtv * u;
    float y = 0.f;
#pragma unroll
    for (int n = 0; n < DS; ++n) {
      const float a = __expf(dtv * Ar[n]);
      h[n] = a * h[n] + du * srow[DR + n];
      y   += h[n] * srow[DR + DS + n];
    }
    float yv = y + u * Dv;
    const float z = xz[((size_t)bb * L + l) * 512 + 256 + d];
    yv *= z / (1.f + __expf(-z));
    ybf[((size_t)bb * L + l) * 256 + d] = (bf16_t)yv;
  }
}

// ---------------------------------------------------------------------------
// K7: out_proj GEMM  ym[65536 x 128] = y[65536 x 256] * wot[256 x 128]
// ---------------------------------------------------------------------------
__global__ void outproj_gemm_kernel(const bf16_t* __restrict__ ybf, const bf16_t* __restrict__ wot,
                                    float* __restrict__ ym)
{
  const int wave = (blockIdx.x * blockDim.x + threadIdx.x) >> 5;
  const int lane = threadIdx.x & 31;
  const int ng = wave & 1, mt = wave >> 1;              // 4096 x 2 groups of 64
  const int m0 = mt * 16, n0 = ng * 64;
  v8f acc[4]; zero_acc<4>(acc);
  gemm_tile_bf16<4>(ybf, 256, wot, 128, m0, n0, 256, lane, acc);
  const int half = lane >> 4, col = lane & 15;
#pragma unroll
  for (int r = 0; r < 8; ++r)
#pragma unroll
    for (int j = 0; j < 4; ++j)
      ym[(size_t)(m0 + r + 8 * half) * 128 + n0 + j * 16 + col] = acc[j][r];
}

// ---------------------------------------------------------------------------
// K8: un-flip + average 4 directions, add h1 residual, transpose to [b][c][l]
// ---------------------------------------------------------------------------
__global__ void combine_kernel(const float* __restrict__ ym, const float* __restrict__ h1,
                               bf16_t* __restrict__ rt)
{
  const int idx = blockIdx.x * 256 + threadIdx.x;       // 4*4096*128
  const int c = idx & 127, l = (idx >> 7) & 4095, b = idx >> 19;
  const int lf = L - 1 - l, cf = 127 - c;
  float v = ym[((size_t)(0 * 4 + b) * L + l ) * 128 + c ]
          + ym[((size_t)(1 * 4 + b) * L + l ) * 128 + cf]
          + ym[((size_t)(2 * 4 + b) * L + lf) * 128 + c ]
          + ym[((size_t)(3 * 4 + b) * L + lf) * 128 + cf];
  v = v * 0.25f + h1[((size_t)b * L + l) * 128 + c];
  rt[((size_t)b * 128 + c) * L + l] = (bf16_t)v;
}

// ---------------------------------------------------------------------------
// K9: nin2 GEMM + BN2 + ReLU -> d_out as [b][o][l]  (== [B,C,H,W] f32)
// ---------------------------------------------------------------------------
__global__ void nin2_gemm_kernel(const bf16_t* __restrict__ W, const bf16_t* __restrict__ rt,
                                 const float* __restrict__ g,  const float* __restrict__ bt,
                                 const float* __restrict__ mn, const float* __restrict__ vr,
                                 float* __restrict__ out)
{
  const int wave = (blockIdx.x * blockDim.x + threadIdx.x) >> 5;
  const int lane = threadIdx.x & 31;
  const int ng = wave & 63, mt = (wave >> 6) & 7, b = wave >> 9;
  const int m0 = mt * 16, n0 = ng * 64;
  v8f acc[4]; zero_acc<4>(acc);
  gemm_tile_bf16<4>(W, 128, rt + (size_t)b * 128 * L, L, m0, n0, 128, lane, acc);
  const int half = lane >> 4, col = lane & 15;
#pragma unroll
  for (int r = 0; r < 8; ++r) {
    const int o = m0 + r + 8 * half;
    const float sc = g[o] * rsqrtf(vr[o] + EPS);
    const float bi = bt[o] - mn[o] * sc;
#pragma unroll
    for (int j = 0; j < 4; ++j) {
      const int l = n0 + j * 16 + col;
      out[((size_t)b * 128 + o) * L + l] = fmaxf(acc[j][r] * sc + bi, 0.f);
    }
  }
}

// ---------------------------------------------------------------------------
extern "C" void kernel_launch(void* const* d_in, const int* in_sizes, int n_in,
                              void* d_out, int out_size, void* d_ws, size_t ws_size,
                              hipStream_t stream)
{
  const float* x         = (const float*)d_in[0];
  const float* nin_w     = (const float*)d_in[1];
  const float* bn1_g     = (const float*)d_in[2];
  const float* bn1_b     = (const float*)d_in[3];
  const float* bn1_m     = (const float*)d_in[4];
  const float* bn1_v     = (const float*)d_in[5];
  const float* in_proj_w = (const float*)d_in[6];
  const float* conv_w    = (const float*)d_in[7];
  const float* conv_b    = (const float*)d_in[8];
  const float* x_proj_w  = (const float*)d_in[9];
  const float* dt_proj_w = (const float*)d_in[10];
  const float* dt_proj_b = (const float*)d_in[11];
  const float* A_log     = (const float*)d_in[12];
  const float* Dp        = (const float*)d_in[13];
  const float* out_proj_w= (const float*)d_in[14];
  const float* nin2_w    = (const float*)d_in[15];
  const float* bn2_g     = (const float*)d_in[16];
  const float* bn2_b     = (const float*)d_in[17];
  const float* bn2_m     = (const float*)d_in[18];
  const float* bn2_v     = (const float*)d_in[19];
  float* out = (float*)d_out;

  char* ws = (char*)d_ws;
  bf16_t* w1   = (bf16_t*)(ws + OFF_W1);
  bf16_t* w2   = (bf16_t*)(ws + OFF_W2);
  bf16_t* wti  = (bf16_t*)(ws + OFF_WTI);
  bf16_t* wxpt = (bf16_t*)(ws + OFF_WXPT);
  bf16_t* wot  = (bf16_t*)(ws + OFF_WOT);
  float*  Aexp = (float*) (ws + OFF_AEXP);
  bf16_t* xb   = (bf16_t*)(ws + OFF_XB);
  float*  h1   = (float*) (ws + OFF_H1);
  bf16_t* xs   = (bf16_t*)(ws + OFF_XS);
  float*  xz   = (float*) (ws + OFF_XZ);
  float*  uc   = (float*) (ws + OFF_UC);
  bf16_t* ucb  = (bf16_t*)(ws + OFF_UCB);
  float*  dbl  = (float*) (ws + OFF_DBL);
  bf16_t* ybf  = (bf16_t*)(ws + OFF_YBF);
  float*  ym   = (float*) (ws + OFF_YM);
  bf16_t* rt   = (bf16_t*)(ws + OFF_RT);
  float*  Pc   = (float*) (ws + OFF_P);
  float*  Sc   = (float*) (ws + OFF_S);
  float*  Hin  = (float*) (ws + OFF_HIN);

  const dim3 blk(256);

  // K0: prep (x->bf16, weight transposes, -exp(A_log))
  {
    const int ntot = 4*128*4096 + 128*128*2 + 128*512 + 256*64 + 256*128 + 256*16;
    prep_kernel<<<dim3((ntot + 255) / 256), blk, 0, stream>>>(
        x, nin_w, nin2_w, in_proj_w, x_proj_w, out_proj_w, A_log,
        xb, w1, w2, wti, wxpt, wot, Aexp);
  }
  // K1: nin1 + BN + ReLU  (4*8*64 waves of 16x64)
  nin1_gemm_kernel<<<dim3(256), blk, 0, stream>>>(w1, xb, bn1_g, bn1_b, bn1_m, bn1_v, h1);
  // K2: 4-direction copies
  build_xs_kernel<<<dim3(16 * 4096 * 128 / 256), blk, 0, stream>>>(h1, xs);
  // K3: in_proj (4096*8 waves of 16x64)
  inproj_gemm_kernel<<<dim3(4096), blk, 0, stream>>>(xs, wti, xz);
  // K4: conv + SiLU
  conv_silu_kernel<<<dim3(16 * 4096 * 256 / 256), blk, 0, stream>>>(xz, conv_w, conv_b, uc, ucb);
  // K5: x_proj (4096 waves of 16x64)
  xproj_gemm_kernel<<<dim3(512), blk, 0, stream>>>(ucb, wxpt, dbl);
  // K6: chunked selective scan (16 seqs x 64 chunks)
  scan_part1_kernel<<<dim3(16 * NC), blk, 0, stream>>>(dbl, uc, Aexp, dt_proj_w, dt_proj_b, Pc, Sc);
  scan_part2_kernel<<<dim3(16 * 256 * 16 / 256), blk, 0, stream>>>(Pc, Sc, Hin);
  scan_part3_kernel<<<dim3(16 * NC), blk, 0, stream>>>(dbl, uc, xz, Aexp, dt_proj_w, dt_proj_b,
                                                       Dp, Hin, ybf);
  // K7: out_proj (4096*2 waves of 16x64)
  outproj_gemm_kernel<<<dim3(1024), blk, 0, stream>>>(ybf, wot, ym);
  // K8: combine directions + residual + transpose
  combine_kernel<<<dim3(4 * 4096 * 128 / 256), blk, 0, stream>>>(ym, h1, rt);
  // K9: nin2 + BN + ReLU
  nin2_gemm_kernel<<<dim3(256), blk, 0, stream>>>(w2, rt, bn2_g, bn2_b, bn2_m, bn2_v, out);
}
